// GraphOrderingPolicy_86311662780921
// MI455X (gfx1250) — compile-verified, matrix operand order checked by reading, not run on hardware
//
#include <hip/hip_runtime.h>

#define NN 2048
#define EMB 128

typedef __attribute__((ext_vector_type(16))) __bf16 v16bf;
typedef __attribute__((ext_vector_type(8)))  float  v8f;

// POD 16-byte chunk (avoid HIP vector-class-in-union issues)
struct U4 { unsigned x, y, z, w; };

union Frag { U4 u[2]; v16bf v; };

__device__ __forceinline__ unsigned short f2bf(float f) {
  unsigned u = __float_as_uint(f);
  u += 0x7fffu + ((u >> 16) & 1u);              // round-to-nearest-even
  return (unsigned short)(u >> 16);
}

// Low 32 bits of a flat pointer into LDS are the LDS byte offset (ISA 10.2 aperture rules)
__device__ __forceinline__ unsigned lds_off(const void* p) {
  return (unsigned)(unsigned long long)p;
}

// CDNA5 async global->LDS copy, 16B per lane, tracked by ASYNCcnt (ISA 08 §4)
__device__ __forceinline__ void async_copy_b128(unsigned lds_dst, const void* gsrc) {
  asm volatile("global_load_async_to_lds_b128 %0, %1, off"
               :: "v"(lds_dst), "v"((unsigned long long)gsrc)
               : "memory");
}
__device__ __forceinline__ void wait_async0() {
  asm volatile("s_wait_asynccnt 0x0" ::: "memory");
}

// ---------------- degree / normalization prep ----------------
__global__ void k_rowsum(const float* __restrict__ adj, float* __restrict__ deg) {
  __shared__ float red[256];
  int row = blockIdx.x;
  const float* r = adj + (size_t)row * NN;
  float s = 0.f;
  for (int j = threadIdx.x; j < NN; j += 256) s += r[j];
  red[threadIdx.x] = s; __syncthreads();
  for (int off = 128; off > 0; off >>= 1) {
    if (threadIdx.x < off) red[threadIdx.x] += red[threadIdx.x + off];
    __syncthreads();
  }
  if (threadIdx.x == 0) deg[row] = red[0];
}

__global__ void k_prep(const float* __restrict__ deg, float* __restrict__ dis,
                       float* __restrict__ dnm) {
  __shared__ float red[256];
  float m = 1.f;
  for (int i = threadIdx.x; i < NN; i += 256) m = fmaxf(m, deg[i]);
  red[threadIdx.x] = m; __syncthreads();
  for (int off = 128; off > 0; off >>= 1) {
    if (threadIdx.x < off) red[threadIdx.x] = fmaxf(red[threadIdx.x], red[threadIdx.x + off]);
    __syncthreads();
  }
  float mx = fmaxf(red[0], 1.f);
  for (int i = threadIdx.x; i < NN; i += 256) {
    float d = deg[i];
    dis[i] = rsqrtf(d + 1.f);       // rowsum(adj+I) = deg+1 >= 1
    dnm[i] = d / mx;
  }
}

// ---------------- GCN layer 1: h = relu((a_norm @ [1,dnorm]) @ W1^T + b1), bf16 out ----------------
__global__ void k_gcn1(const float* __restrict__ adj, const float* __restrict__ dis,
                       const float* __restrict__ dnm, const float* __restrict__ w1,
                       const float* __restrict__ b1, unsigned short* __restrict__ h_bf) {
  __shared__ float r0[256], r1[256];
  __shared__ float y0s, y1s;
  int row = blockIdx.x;
  const float* ar = adj + (size_t)row * NN;
  float s0 = 0.f, s1 = 0.f;
  for (int j = threadIdx.x; j < NN; j += 256) {
    float a = ar[j] * dis[j];
    s0 += a;
    s1 += a * dnm[j];
  }
  r0[threadIdx.x] = s0; r1[threadIdx.x] = s1; __syncthreads();
  for (int off = 128; off > 0; off >>= 1) {
    if (threadIdx.x < off) { r0[threadIdx.x] += r0[threadIdx.x + off]; r1[threadIdx.x] += r1[threadIdx.x + off]; }
    __syncthreads();
  }
  if (threadIdx.x == 0) {
    float di = dis[row];
    float c0 = r0[0] + di;             // + I diagonal contribution
    float c1 = r1[0] + di * dnm[row];
    y0s = di * c0; y1s = di * c1;
  }
  __syncthreads();
  if (threadIdx.x < EMB) {
    int e = threadIdx.x;
    float t = y0s * w1[2 * e] + y1s * w1[2 * e + 1] + b1[e];
    h_bf[(size_t)row * EMB + e] = f2bf(fmaxf(t, 0.f));
  }
}

__global__ void k_cvt(const float* __restrict__ in, unsigned short* __restrict__ out, int n) {
  int i = blockIdx.x * 256 + threadIdx.x;
  if (i < n) out[i] = f2bf(in[i]);
}

// ---------------- generic bf16 WMMA GEMM: C[M][128] = A[M][K] @ Bt[128][K]^T (+bias) ----------------
// Double-buffered staging via GLOBAL_LOAD_ASYNC_TO_LDS_B128 (ASYNCcnt pipeline).
__global__ __launch_bounds__(256) void k_gemm_bf16(
    const unsigned short* __restrict__ A, const unsigned short* __restrict__ Bt,
    const float* __restrict__ bias, float* __restrict__ outf,
    unsigned short* __restrict__ outb, int K) {
  __shared__ __align__(16) unsigned short sA[2][128 * 32];
  __shared__ __align__(16) unsigned short sB[2][128 * 32];   // [n][k]
  const int tid = threadIdx.x, lane = tid & 31, wave = tid >> 5;
  const int m0 = wave * 16;
  const int mg = blockIdx.x * 128;

  // this thread's two staging chunks (row r, k-offset ko within the 128x32 tile)
  const int r0s = tid >> 2,          ko0 = (tid & 3) * 8;
  const int r1s = (tid + 256) >> 2,  ko1 = ((tid + 256) & 3) * 8;

  auto stage = [&](int b, int kk) {
    async_copy_b128(lds_off(&sA[b][r0s * 32 + ko0]), &A[(size_t)(mg + r0s) * K + kk + ko0]);
    async_copy_b128(lds_off(&sA[b][r1s * 32 + ko1]), &A[(size_t)(mg + r1s) * K + kk + ko1]);
    async_copy_b128(lds_off(&sB[b][r0s * 32 + ko0]), &Bt[(size_t)r0s * K + kk + ko0]);
    async_copy_b128(lds_off(&sB[b][r1s * 32 + ko1]), &Bt[(size_t)r1s * K + kk + ko1]);
  };

  v8f acc[8] = {};
  int buf = 0;
  stage(0, 0);
  for (int kk = 0; kk < K; kk += 32) {
    wait_async0();        // my copies into sX[buf] have landed in LDS
    __syncthreads();      // everyone's copies landed; prev buffer fully consumed
    if (kk + 32 < K) stage(buf ^ 1, kk + 32);   // prefetch next tile

    Frag fa;
    int rr = (lane & 15) * 32;
    int ah = (lane >> 4) * 8;
    fa.u[0] = *(const U4*)&sA[buf][m0 * 32 + rr + ah];        // ds_read_b128
    fa.u[1] = *(const U4*)&sA[buf][m0 * 32 + rr + 16 + ah];
    #pragma unroll
    for (int nt = 0; nt < 8; ++nt) {
      Frag fb;
      int nb = (nt * 16 + (lane & 15)) * 32;
      fb.u[0] = *(const U4*)&sB[buf][nb + ah];
      fb.u[1] = *(const U4*)&sB[buf][nb + 16 + ah];
      acc[nt] = __builtin_amdgcn_wmma_f32_16x16x32_bf16(
          false, fa.v, false, fb.v, (short)0, acc[nt], false, false);
    }
    buf ^= 1;
  }
  int mrow = mg + m0 + ((lane >> 4) ? 8 : 0);
  int nc0 = lane & 15;
  #pragma unroll
  for (int nt = 0; nt < 8; ++nt) {
    int col = nt * 16 + nc0;
    float b = bias ? bias[col] : 0.f;
    #pragma unroll
    for (int r = 0; r < 8; ++r) {
      float val = acc[nt][r] + b;
      size_t idx = (size_t)(mrow + r) * 128 + col;
      if (outf) outf[idx] = val;
      if (outb) outb[idx] = f2bf(val);
    }
  }
}

// ---------------- emb = a_norm(on-the-fly bf16) @ hw + b2; a_norm = dis_i*(adj+I)*dis_j ----------------
__global__ __launch_bounds__(256) void k_spmm_wmma(
    const float* __restrict__ adj, const float* __restrict__ dis,
    const unsigned short* __restrict__ hwb, const float* __restrict__ bias,
    float* __restrict__ embf, unsigned short* __restrict__ embb) {
  __shared__ __align__(16) unsigned short sA[128 * 32];
  __shared__ __align__(16) unsigned short sB[128 * 32];   // [n][k]
  const int tid = threadIdx.x, lane = tid & 31, wave = tid >> 5;
  const int m0 = wave * 16;
  const int mg = blockIdx.x * 128;
  v8f acc[8] = {};
  for (int kk = 0; kk < NN; kk += 32) {
    // A tile: rows mg..mg+127, cols kk..kk+31 of a_norm, quantized bf16
    #pragma unroll
    for (int it = 0; it < 2; ++it) {
      int t = tid + it * 256;
      int r = t >> 2, ko = (t & 3) * 8;
      int gi = mg + r;
      float di = dis[gi];
      const float* arow = adj + (size_t)gi * NN + kk + ko;
      __builtin_prefetch(arow + 32, 0, 1);               // global_prefetch_b8: next K-tile
      __align__(16) unsigned short tmp[8];
      #pragma unroll
      for (int e = 0; e < 8; ++e) {
        int gj = kk + ko + e;
        float a = arow[e] + ((gi == gj) ? 1.f : 0.f);
        tmp[e] = f2bf(a * di * dis[gj]);
      }
      *(U4*)&sA[r * 32 + ko] = *(const U4*)tmp;
    }
    // B tile transposed into LDS: sB[n][k] = hw[kk+k][n]
    #pragma unroll
    for (int it = 0; it < 2; ++it) {
      int t = tid + it * 256;
      int kl = t >> 4, no = (t & 15) * 8;
      U4 q = *(const U4*)&hwb[(size_t)(kk + kl) * 128 + no];
      const unsigned short* qp = (const unsigned short*)&q;
      #pragma unroll
      for (int e = 0; e < 8; ++e) sB[(no + e) * 32 + kl] = qp[e];
    }
    __syncthreads();
    Frag fa;
    int rr = (lane & 15) * 32;
    int ah = (lane >> 4) * 8;
    fa.u[0] = *(const U4*)&sA[m0 * 32 + rr + ah];
    fa.u[1] = *(const U4*)&sA[m0 * 32 + rr + 16 + ah];
    #pragma unroll
    for (int nt = 0; nt < 8; ++nt) {
      Frag fb;
      int nb = (nt * 16 + (lane & 15)) * 32;
      fb.u[0] = *(const U4*)&sB[nb + ah];
      fb.u[1] = *(const U4*)&sB[nb + 16 + ah];
      acc[nt] = __builtin_amdgcn_wmma_f32_16x16x32_bf16(
          false, fa.v, false, fb.v, (short)0, acc[nt], false, false);
    }
    __syncthreads();
  }
  int mrow = mg + m0 + ((lane >> 4) ? 8 : 0);
  int nc0 = lane & 15;
  #pragma unroll
  for (int nt = 0; nt < 8; ++nt) {
    int col = nt * 16 + nc0;
    float b = bias[col];
    #pragma unroll
    for (int r = 0; r < 8; ++r) {
      float val = acc[nt][r] + b;
      size_t idx = (size_t)(mrow + r) * 128 + col;
      embf[idx] = val;
      embb[idx] = f2bf(val);
    }
  }
}

// ---------------- persistent single-WGP pointer decoder (2048 sequential steps) ----------------
__global__ __launch_bounds__(1024) void k_decoder(
    const float* __restrict__ whe, const float* __restrict__ emb,
    const float* __restrict__ Ws, const float* __restrict__ bs,
    const float* __restrict__ vvec, const float* __restrict__ state0,
    const float* __restrict__ wi, const float* __restrict__ wh,
    const float* __restrict__ bi, const float* __restrict__ bh,
    float* __restrict__ out) {
  __shared__ float s_state[128], s_u[128], s_v[128], s_x[128];
  __shared__ float s_gi[384], s_gh[384];
  __shared__ float s_logit[NN];
  __shared__ unsigned char s_mask[NN];
  __shared__ float s_red[1024], s_red2[1024];
  __shared__ int s_idx[1024];
  __shared__ float s_maxv, s_lp, s_ent;
  __shared__ int s_aidx;

  const int tid = threadIdx.x;
  if (tid < 128) { s_state[tid] = state0[tid]; s_v[tid] = vvec[tid]; }
  for (int i = tid; i < NN; i += 1024) s_mask[i] = 1;
  if (tid == 0) { s_lp = 0.f; s_ent = 0.f; }
  __syncthreads();

  for (int step = 0; step < NN; ++step) {
    // u = Ws @ state + bs
    if (tid < 128) {
      const float4* wr = (const float4*)(Ws + tid * 128);
      float u = bs[tid];
      #pragma unroll 4
      for (int q = 0; q < 32; ++q) {
        float4 w = wr[q];
        const float* st = &s_state[q * 4];
        u += w.x * st[0] + w.y * st[1] + w.z * st[2] + w.w * st[3];
      }
      s_u[tid] = u;
    }
    __syncthreads();
    // logits_i = v . tanh(Wh_emb_i + u), masked
    for (int i = tid; i < NN; i += 1024) {
      const float4* hr = (const float4*)(whe + (size_t)i * 128);
      float acc = 0.f;
      for (int q = 0; q < 32; ++q) {
        float4 w = hr[q];
        acc += tanhf(w.x + s_u[q * 4 + 0]) * s_v[q * 4 + 0];
        acc += tanhf(w.y + s_u[q * 4 + 1]) * s_v[q * 4 + 1];
        acc += tanhf(w.z + s_u[q * 4 + 2]) * s_v[q * 4 + 2];
        acc += tanhf(w.w + s_u[q * 4 + 3]) * s_v[q * 4 + 3];
      }
      s_logit[i] = s_mask[i] ? acc : -1e9f;
    }
    __syncthreads();
    // argmax (first-index tie break)
    {
      float bv = -3e38f; int bim = 0;
      for (int i = tid; i < NN; i += 1024) {
        float x = s_logit[i];
        if (x > bv) { bv = x; bim = i; }
      }
      s_red[tid] = bv; s_idx[tid] = bim;
      __syncthreads();
      for (int off = 512; off > 0; off >>= 1) {
        if (tid < off) {
          float a = s_red[tid], b = s_red[tid + off];
          int ia = s_idx[tid], ib = s_idx[tid + off];
          if (b > a || (b == a && ib < ia)) { s_red[tid] = b; s_idx[tid] = ib; }
        }
        __syncthreads();
      }
      if (tid == 0) { s_maxv = s_red[0]; s_aidx = s_idx[0]; }
      __syncthreads();
    }
    // log-softmax stats: Z = sum exp(l-max), S2 = sum exp(l-max)*(l-max)
    {
      float mv = s_maxv, S1 = 0.f, S2 = 0.f;
      for (int i = tid; i < NN; i += 1024) {
        float d = s_logit[i] - mv;
        float p = expf(d);
        S1 += p; S2 += p * d;
      }
      s_red[tid] = S1; s_red2[tid] = S2;
      __syncthreads();
      for (int off = 512; off > 0; off >>= 1) {
        if (tid < off) { s_red[tid] += s_red[tid + off]; s_red2[tid] += s_red2[tid + off]; }
        __syncthreads();
      }
      if (tid == 0) {
        float Z = s_red[0], lz = logf(Z);
        s_lp += -lz;                       // logp[argmax] = -log Z
        s_ent += lz - s_red2[0] / Z;       // ent -= sum p*logp
        s_mask[s_aidx] = 0;
        out[step] = (float)s_aidx;
      }
      __syncthreads();
    }
    // GRU cell: state = (1-z)*tanh(inn + r*hn) + z*state
    if (tid < 128) s_x[tid] = emb[(size_t)s_aidx * 128 + tid];
    __syncthreads();
    if (tid < 384) {
      const float4* wir = (const float4*)(wi + tid * 128);
      const float4* whr = (const float4*)(wh + tid * 128);
      float a = bi[tid], b = bh[tid];
      for (int q = 0; q < 32; ++q) {
        float4 x = wir[q]; float4 y = whr[q];
        const float* xp = &s_x[q * 4]; const float* sp = &s_state[q * 4];
        a += x.x * xp[0] + x.y * xp[1] + x.z * xp[2] + x.w * xp[3];
        b += y.x * sp[0] + y.y * sp[1] + y.z * sp[2] + y.w * sp[3];
      }
      s_gi[tid] = a; s_gh[tid] = b;
    }
    __syncthreads();
    if (tid < 128) {
      float r = 1.f / (1.f + expf(-(s_gi[tid] + s_gh[tid])));
      float z = 1.f / (1.f + expf(-(s_gi[128 + tid] + s_gh[128 + tid])));
      float ng = tanhf(s_gi[256 + tid] + r * s_gh[256 + tid]);
      s_state[tid] = (1.f - z) * ng + z * s_state[tid];
    }
    __syncthreads();
  }
  if (tid == 0) { out[NN] = s_lp; out[NN + 1] = s_ent; }
}

extern "C" void kernel_launch(void* const* d_in, const int* in_sizes, int n_in,
                              void* d_out, int out_size, void* d_ws, size_t ws_size,
                              hipStream_t stream) {
  const float* adj = (const float*)d_in[0];
  const float* g1w = (const float*)d_in[1];
  const float* g1b = (const float*)d_in[2];
  const float* g2w = (const float*)d_in[3];
  const float* g2b = (const float*)d_in[4];
  const float* Wh  = (const float*)d_in[5];
  const float* Ws  = (const float*)d_in[6];
  const float* bs  = (const float*)d_in[7];
  const float* v   = (const float*)d_in[8];
  const float* st0 = (const float*)d_in[9];
  const float* gwi = (const float*)d_in[10];
  const float* gwh = (const float*)d_in[11];
  const float* gbi = (const float*)d_in[12];
  const float* gbh = (const float*)d_in[13];
  float* out = (float*)d_out;

  char* ws = (char*)d_ws;
  size_t off = 0;
  auto alloc = [&](size_t bytes) -> void* {
    void* p = ws + off;
    off += bytes;
    off = (off + 255) & ~(size_t)255;
    return p;
  };
  float* deg = (float*)alloc(NN * 4);
  float* dis = (float*)alloc(NN * 4);
  float* dnm = (float*)alloc(NN * 4);
  unsigned short* h_bf   = (unsigned short*)alloc((size_t)NN * EMB * 2);
  unsigned short* g2w_bf = (unsigned short*)alloc(EMB * EMB * 2);
  unsigned short* wh_bf  = (unsigned short*)alloc(EMB * EMB * 2);
  unsigned short* hw_bf  = (unsigned short*)alloc((size_t)NN * EMB * 2);
  unsigned short* emb_bf = (unsigned short*)alloc((size_t)NN * EMB * 2);
  float* emb_f = (float*)alloc((size_t)NN * EMB * 4);
  float* whe_f = (float*)alloc((size_t)NN * EMB * 4);

  k_rowsum<<<NN, 256, 0, stream>>>(adj, deg);
  k_prep<<<1, 256, 0, stream>>>(deg, dis, dnm);
  k_gcn1<<<NN, 256, 0, stream>>>(adj, dis, dnm, g1w, g1b, h_bf);
  k_cvt<<<(EMB * EMB + 255) / 256, 256, 0, stream>>>(g2w, g2w_bf, EMB * EMB);
  k_cvt<<<(EMB * EMB + 255) / 256, 256, 0, stream>>>(Wh, wh_bf, EMB * EMB);
  // hw = h @ gcn2_w^T   (Bt = gcn2_w row-major == W^T transposed)
  k_gemm_bf16<<<NN / 128, 256, 0, stream>>>(h_bf, g2w_bf, nullptr, nullptr, hw_bf, EMB);
  // emb = a_norm @ hw + gcn2_b
  k_spmm_wmma<<<NN / 128, 256, 0, stream>>>(adj, dis, hw_bf, g2b, emb_f, emb_bf);
  // Wh_emb = emb @ Wh^T
  k_gemm_bf16<<<NN / 128, 256, 0, stream>>>(emb_bf, wh_bf, nullptr, whe_f, nullptr, EMB);
  // sequential greedy decode on one WGP
  k_decoder<<<1, 1024, 0, stream>>>(whe_f, emb_f, Ws, bs, v, st0, gwi, gwh, gbi, gbh, out);
}
